// ExpertWiseSparseGLU_7327214207125
// MI455X (gfx1250) — matrix operand order repaired and checked by reading, not compiled, not run on hardware
//
#include <hip/hip_runtime.h>
#include <hip/hip_bf16.h>
#include <math.h>

// Problem constants (match reference)
#define E_ 8
#define H_ 1024
#define F_ 2048
#define T_ 8192
#define TPE (T_ / E_)      // 1024 tokens per expert

// Tiling
#define BM  64             // token rows per block
#define BN1 128            // stage-1 F columns per block
#define BN2 128            // stage-2 H columns per block
#define BK  32             // K chunk (bf16 WMMA K)
#define LDA 40             // padded LDS row stride in halves (80B, 16B-aligned, bank-spread)
#define NK1 (H_ / BK)      // 32 K-chunks in stage 1
#define NK2 (F_ / BK)      // 64 K-chunks in stage 2

typedef __attribute__((ext_vector_type(16))) __bf16 v16bf;
typedef __attribute__((ext_vector_type(2)))  __bf16 v2bf;
typedef __attribute__((ext_vector_type(8)))  float  v8f;
typedef __attribute__((ext_vector_type(4)))  unsigned int v4u;
typedef __attribute__((ext_vector_type(8)))  int v8i;
typedef __attribute__((ext_vector_type(4)))  int v4i;

union FragAB { v16bf v; uint4 q[2]; unsigned short h[16]; };
union FragC  { v8f   v; float f[8]; };

// ---- feature detection (keeps the build green on either toolchain) ----
#if __has_builtin(__builtin_amdgcn_tensor_load_to_lds) && __has_builtin(__builtin_amdgcn_s_wait_tensorcnt)
#define HAVE_TDM 1
#else
#define HAVE_TDM 0
#endif
#if __has_builtin(__builtin_amdgcn_global_load_async_to_lds_b128) && __has_builtin(__builtin_amdgcn_s_wait_asynccnt)
#define HAVE_ASYNC 1
#else
#define HAVE_ASYNC 0
#endif
#if HAVE_TDM && __has_include(<hip/amd_detail/amd_gfx1250_TDM.h>)
#define TDM_SIX_ARGS 1
#else
#define TDM_SIX_ARGS 0
#endif
#if __has_builtin(__builtin_amdgcn_cvt_pk_bf16_f32)
#define HAVE_PKBF 1
#else
#define HAVE_PKBF 0
#endif

// fp32 -> bf16, round-to-nearest (ties away): 2 VALU ops
__device__ __forceinline__ unsigned short f2bf(float f) {
  return (unsigned short)((__float_as_uint(f) + 0x8000u) >> 16);
}

// pack two fp32 -> bf16x2: hardware pk-convert if present, else one v_perm_b32
// selecting the high halves of the two rounded words (3 VALU ops per pair).
__device__ __forceinline__ unsigned int f2bf_pk(float a, float b) {
#if HAVE_PKBF
  union { v2bf v; unsigned int u; } cv;
  cv.v = __builtin_amdgcn_cvt_pk_bf16_f32(a, b);
  return cv.u;
#else
  return __builtin_amdgcn_perm(__float_as_uint(b) + 0x8000u,   // bytes 7:4
                               __float_as_uint(a) + 0x8000u,   // bytes 3:0
                               0x07060302u);                   // {b.hi16, a.hi16}
#endif
}

__device__ __forceinline__ float gelu_erf(float x) {
  return 0.5f * x * (1.0f + erff(x * 0.70710678118654752f));
}

#if HAVE_TDM
// TDM: DMA a [tile_rows x tile_k] bf16 tile (row stride tensor_w elements) into
// LDS at lds_byte_addr, inserting 16B of padding after every 64B row so rows
// land at the kernel's LDA(=40 half) stride.  D# layout per CDNA5 ISA ch.8.
__device__ __forceinline__ void tdm_load_tile_bf16(const unsigned short* gsrc,
                                                   unsigned lds_byte_addr,
                                                   unsigned tile_k, unsigned tile_rows,
                                                   unsigned tensor_w, unsigned tensor_h) {
  unsigned long long ga = (unsigned long long)gsrc;
  v4u g0;
  g0.x = 1u;                                                // count=1 valid descriptor
  g0.y = lds_byte_addr;                                     // lds_addr [63:32]
  g0.z = (unsigned)(ga & 0xFFFFFFFFu);                      // global_addr lo
  g0.w = (unsigned)((ga >> 32) & 0x01FFFFFFu) | (2u << 30); // global_addr hi | type=2
  v8i g1;
  // data_size=2B (code 1), pad_enable, pad_interval=3 (16 DW = 64B row),
  // pad_amount=3 (4 DW = 16B pad)  -> LDS row stride 80B = LDA halves
  g1[0] = (int)((1u << 16) | (1u << 20) | (3u << 22) | (3u << 25));
  g1[1] = (int)((tensor_w & 0xFFFFu) << 16);                // tensor_dim0 lo16
  g1[2] = (int)((tensor_w >> 16) | ((tensor_h & 0xFFFFu) << 16));
  g1[3] = (int)((tensor_h >> 16) | (tile_k << 16));         // tile_dim0
  g1[4] = (int)tile_rows;                                   // tile_dim1 (tile_dim2=0)
  g1[5] = (int)tensor_w;                                    // tensor_dim0_stride lo32
  g1[6] = 0;
  g1[7] = 0;
  v4i gz = {0, 0, 0, 0};
#if TDM_SIX_ARGS
  v8i gz8 = {0, 0, 0, 0, 0, 0, 0, 0};
  __builtin_amdgcn_tensor_load_to_lds(g0, g1, gz, gz, gz8, 0);
#else
  __builtin_amdgcn_tensor_load_to_lds(g0, g1, gz, gz, 0);
#endif
}
#endif

// -------- Stage 1: g = gelu(x @ w1^T) * (x @ v1^T), bf16 output [T, F] --------
// Block tile 64(M) x 128(F); 8 waves, each wave owns a 32x32 subtile of both GEMMs.
// Ping-pong LDS: global loads of chunk k+1 issue before the WMMAs of chunk k.
__global__ __launch_bounds__(256)
void moe_glu_stage1(const float* __restrict__ x,
                    const float* __restrict__ w1,
                    const float* __restrict__ v1,
                    unsigned short* __restrict__ gout) {
  __shared__ __align__(16) unsigned short sA[2][BM  * LDA];
  __shared__ __align__(16) unsigned short sW[2][BN1 * LDA];
  __shared__ __align__(16) unsigned short sV[2][BN1 * LDA];

  const int f0 = blockIdx.x * BN1;
  const int t0 = blockIdx.y * BM;
  const int e  = t0 / TPE;
  const float* w1e = w1 + (size_t)e * F_ * H_;
  const float* v1e = v1 + (size_t)e * F_ * H_;

  const int tid   = threadIdx.x;
  const int lane  = tid & 31;
  const int wave  = tid >> 5;
  const int mg    = wave >> 2;       // 0..1 : 32-row M group
  const int ng    = wave & 3;        // 0..3 : 32-col N group
  const int lhalf = lane >> 4;
  const int l16   = lane & 15;

  // cooperative-load indices
  const int lr = tid >> 2;           // A: row 0..63
  const int ls = (tid & 3) * 8;      // A: k segment
  const int wr = tid >> 1;           // W/V: row 0..127
  const int wc = (tid & 1) * 16;     // W/V: k segment (16 elems)

  float ra[8], rw[16], rv[16];       // staged chunk (registers)

  auto load_chunk = [&](int k0) {
    const float* sx = x + (size_t)(t0 + lr) * H_ + k0 + ls;
    #pragma unroll
    for (int j = 0; j < 8; ++j) ra[j] = sx[j];
    const float* swp = w1e + (size_t)(f0 + wr) * H_ + k0 + wc;
    const float* svp = v1e + (size_t)(f0 + wr) * H_ + k0 + wc;
    #pragma unroll
    for (int j = 0; j < 16; ++j) { rw[j] = swp[j]; rv[j] = svp[j]; }
    if (k0 + BK < H_) {              // speculative prefetch one more chunk ahead
      __builtin_prefetch(swp + BK, 0, 0);
      __builtin_prefetch(svp + BK, 0, 0);
    }
  };
  auto store_chunk = [&](int p) {
    uint4 qa;
    qa.x = f2bf_pk(ra[0], ra[1]); qa.y = f2bf_pk(ra[2], ra[3]);
    qa.z = f2bf_pk(ra[4], ra[5]); qa.w = f2bf_pk(ra[6], ra[7]);
    *(uint4*)&sA[p][lr * LDA + ls] = qa;
    uint4 qw0, qw1, qv0, qv1;
    qw0.x = f2bf_pk(rw[0], rw[1]);   qw0.y = f2bf_pk(rw[2], rw[3]);
    qw0.z = f2bf_pk(rw[4], rw[5]);   qw0.w = f2bf_pk(rw[6], rw[7]);
    qw1.x = f2bf_pk(rw[8], rw[9]);   qw1.y = f2bf_pk(rw[10], rw[11]);
    qw1.z = f2bf_pk(rw[12], rw[13]); qw1.w = f2bf_pk(rw[14], rw[15]);
    qv0.x = f2bf_pk(rv[0], rv[1]);   qv0.y = f2bf_pk(rv[2], rv[3]);
    qv0.z = f2bf_pk(rv[4], rv[5]);   qv0.w = f2bf_pk(rv[6], rv[7]);
    qv1.x = f2bf_pk(rv[8], rv[9]);   qv1.y = f2bf_pk(rv[10], rv[11]);
    qv1.z = f2bf_pk(rv[12], rv[13]); qv1.w = f2bf_pk(rv[14], rv[15]);
    *(uint4*)&sW[p][wr * LDA + wc]     = qw0;
    *(uint4*)&sW[p][wr * LDA + wc + 8] = qw1;
    *(uint4*)&sV[p][wr * LDA + wc]     = qv0;
    *(uint4*)&sV[p][wr * LDA + wc + 8] = qv1;
  };

  FragC acc1[2][2], acc2[2][2];
  #pragma unroll
  for (int sm = 0; sm < 2; ++sm)
    #pragma unroll
    for (int sn = 0; sn < 2; ++sn)
      #pragma unroll
      for (int j = 0; j < 8; ++j) { acc1[sm][sn].f[j] = 0.f; acc2[sm][sn].f[j] = 0.f; }

  // pipeline prologue
  load_chunk(0);
  store_chunk(0);
  __syncthreads();

  int p = 0;
  for (int kc = 0; kc < NK1; ++kc) {
    const bool more = (kc + 1) < NK1;
    if (more) load_chunk((kc + 1) * BK);     // global loads overlap the WMMAs below

    FragAB a[2], bw[2], bv[2];
    #pragma unroll
    for (int sm = 0; sm < 2; ++sm) {
      const unsigned short* ap = &sA[p][(mg * 32 + sm * 16 + l16) * LDA + lhalf * 16];
      a[sm].q[0] = *(const uint4*)(ap);
      a[sm].q[1] = *(const uint4*)(ap + 8);
    }
    #pragma unroll
    for (int sn = 0; sn < 2; ++sn) {
      const unsigned short* wp = &sW[p][(ng * 32 + sn * 16 + l16) * LDA + lhalf * 16];
      bw[sn].q[0] = *(const uint4*)(wp);
      bw[sn].q[1] = *(const uint4*)(wp + 8);
      const unsigned short* vp = &sV[p][(ng * 32 + sn * 16 + l16) * LDA + lhalf * 16];
      bv[sn].q[0] = *(const uint4*)(vp);
      bv[sn].q[1] = *(const uint4*)(vp + 8);
    }

    #pragma unroll
    for (int sm = 0; sm < 2; ++sm)
      #pragma unroll
      for (int sn = 0; sn < 2; ++sn) {
        acc1[sm][sn].v = __builtin_amdgcn_wmma_f32_16x16x32_bf16(
            false, a[sm].v, false, bw[sn].v, (short)0, acc1[sm][sn].v, false, false);
        acc2[sm][sn].v = __builtin_amdgcn_wmma_f32_16x16x32_bf16(
            false, a[sm].v, false, bv[sn].v, (short)0, acc2[sm][sn].v, false, false);
      }

    if (more) store_chunk(1 - p);            // convert + fill the other buffer
    __syncthreads();
    p ^= 1;
  }

  // ---- epilogue: exact GELU gating, bf16 store ----
  #pragma unroll
  for (int sm = 0; sm < 2; ++sm) {
    const int mbase = t0 + mg * 32 + sm * 16 + lhalf * 8;
    #pragma unroll
    for (int sn = 0; sn < 2; ++sn) {
      const int n = f0 + ng * 32 + sn * 16 + l16;
      #pragma unroll
      for (int r = 0; r < 8; ++r) {
        const float gg = gelu_erf(acc1[sm][sn].f[r]) * acc2[sm][sn].f[r];
        gout[(size_t)(mbase + r) * F_ + n] = f2bf(gg);
      }
    }
  }
}

// -------- Stage 2: out = g @ w2 (per expert), fp32 output [T, H] --------
// Block tile 64(M) x 128(H).  A tile (already bf16) streamed by the Tensor Data
// Mover into the ping-pong buffer, overlapping the WMMAs on the live buffer.
__global__ __launch_bounds__(256)
void moe_glu_stage2(const unsigned short* __restrict__ g,
                    const float* __restrict__ w2,
                    float* __restrict__ out) {
  __shared__ __align__(16) unsigned short sA[2][BM  * LDA];
  __shared__ __align__(16) unsigned short sB[2][BN2 * LDA];

  const int h0 = blockIdx.x * BN2;
  const int t0 = blockIdx.y * BM;
  const int e  = t0 / TPE;
  const float* w2e = w2 + (size_t)e * F_ * H_;

  const int tid   = threadIdx.x;
  const int lane  = tid & 31;
  const int wave  = tid >> 5;
  const int mg    = wave >> 2;
  const int ng    = wave & 3;
  const int lhalf = lane >> 4;
  const int l16   = lane & 15;

  const int kk2 = (tid >> 4) * 2;    // B: k row pair 0,2,...,30
  const int nsg = (tid & 15) * 8;    // B: 8-col segment
  const int lr = tid >> 2;           // A fallback copy indices
  const int ls = (tid & 3) * 8;
#if HAVE_TDM
  const unsigned sA_base =
      (unsigned)(unsigned long long)(__attribute__((address_space(3))) unsigned short*)sA;
#endif

  float rb0[8], rb1[8];              // staged w2 chunk (two adjacent k rows)
  uint4 rA;                          // staged A chunk (fallback paths)

  auto issue_A = [&](int p, int k0) {
#if HAVE_TDM
    if (wave == 0)
      tdm_load_tile_bf16(g + (size_t)t0 * F_ + k0,
                         sA_base + (unsigned)(p * BM * LDA * 2), BK, BM, F_, T_);
#elif HAVE_ASYNC
    __builtin_amdgcn_global_load_async_to_lds_b128(
        (__attribute__((address_space(1))) const void*)(g + (size_t)(t0 + lr) * F_ + k0 + ls),
        (__attribute__((address_space(3))) void*)&sA[p][lr * LDA + ls], 0, 0);
#else
    rA = *(const uint4*)(g + (size_t)(t0 + lr) * F_ + k0 + ls);
#endif
  };
  auto commit_A = [&](int p) {       // make A tile of buffer p ready (pre-barrier)
#if HAVE_TDM
    if (wave == 0) __builtin_amdgcn_s_wait_tensorcnt(0);
#elif HAVE_ASYNC
    __builtin_amdgcn_s_wait_asynccnt(0);
#else
    *(uint4*)&sA[p][lr * LDA + ls] = rA;
#endif
  };
  auto load_B = [&](int k0) {
    const float* src0 = w2e + (size_t)(k0 + kk2) * H_ + h0 + nsg;
    const float* src1 = src0 + H_;
    #pragma unroll
    for (int j = 0; j < 8; ++j) { rb0[j] = src0[j]; rb1[j] = src1[j]; }
    if (k0 + BK < F_) __builtin_prefetch(src0 + (size_t)BK * H_, 0, 0);
  };
  auto store_B = [&](int p) {        // pack (k,k+1) pairs, transpose-store as b32
    #pragma unroll
    for (int j = 0; j < 8; ++j)
      *(unsigned int*)&sB[p][(nsg + j) * LDA + kk2] = f2bf_pk(rb0[j], rb1[j]);
  };

  FragC acc[2][2];
  #pragma unroll
  for (int sm = 0; sm < 2; ++sm)
    #pragma unroll
    for (int sn = 0; sn < 2; ++sn)
      #pragma unroll
      for (int j = 0; j < 8; ++j) acc[sm][sn].f[j] = 0.f;

  // pipeline prologue
  issue_A(0, 0);
  load_B(0);
  store_B(0);
  commit_A(0);
  __syncthreads();

  int p = 0;
  for (int kc = 0; kc < NK2; ++kc) {
    const bool more = (kc + 1) < NK2;
    if (more) {
      issue_A(1 - p, (kc + 1) * BK);       // DMA next A tile while we compute
      load_B((kc + 1) * BK);
    }

    FragAB a[2], b[2];
    #pragma unroll
    for (int sm = 0; sm < 2; ++sm) {
      const unsigned short* ap = &sA[p][(mg * 32 + sm * 16 + l16) * LDA + lhalf * 16];
      a[sm].q[0] = *(const uint4*)(ap);
      a[sm].q[1] = *(const uint4*)(ap + 8);
    }
    #pragma unroll
    for (int sn = 0; sn < 2; ++sn) {
      const unsigned short* bp = &sB[p][(ng * 32 + sn * 16 + l16) * LDA + lhalf * 16];
      b[sn].q[0] = *(const uint4*)(bp);
      b[sn].q[1] = *(const uint4*)(bp + 8);
    }

    #pragma unroll
    for (int sm = 0; sm < 2; ++sm)
      #pragma unroll
      for (int sn = 0; sn < 2; ++sn)
        acc[sm][sn].v = __builtin_amdgcn_wmma_f32_16x16x32_bf16(
            false, a[sm].v, false, b[sn].v, (short)0, acc[sm][sn].v, false, false);

    if (more) {
      store_B(1 - p);
      commit_A(1 - p);
    }
    __syncthreads();
    p ^= 1;
  }

  #pragma unroll
  for (int sm = 0; sm < 2; ++sm) {
    const int mbase = t0 + mg * 32 + sm * 16 + lhalf * 8;
    #pragma unroll
    for (int sn = 0; sn < 2; ++sn) {
      const int n = h0 + ng * 32 + sn * 16 + l16;
      #pragma unroll
      for (int r = 0; r < 8; ++r)
        out[(size_t)(mbase + r) * H_ + n] = acc[sm][sn].f[r];
    }
  }
}

extern "C" void kernel_launch(void* const* d_in, const int* in_sizes, int n_in,
                              void* d_out, int out_size, void* d_ws, size_t ws_size,
                              hipStream_t stream) {
  (void)in_sizes; (void)n_in; (void)out_size; (void)ws_size;
  const float* x  = (const float*)d_in[0];
  // d_in[1] = expert_idx: tokens are pre-sorted with an exact equal split, so
  // expert = token / TPE; the index array is not needed.
  const float* w1 = (const float*)d_in[2];
  const float* v1 = (const float*)d_in[3];
  const float* w2 = (const float*)d_in[4];
  unsigned short* g = (unsigned short*)d_ws;   // bf16 [T, F] = 32 MB scratch
  float* outp = (float*)d_out;

  dim3 blk(256);
  dim3 g1(F_ / BN1, T_ / BM);   // 16 x 128 blocks
  moe_glu_stage1<<<g1, blk, 0, stream>>>(x, w1, v1, g);
  dim3 g2(H_ / BN2, T_ / BM);   // 8 x 128 blocks
  moe_glu_stage2<<<g2, blk, 0, stream>>>(g, w2, outp);
}
